// HOPE_23210003268113
// MI455X (gfx1250) — matrix operand order repaired and checked by reading, not compiled, not run on hardware
//
#include <hip/hip_runtime.h>
#include <hip/hip_bf16.h>
#include <math.h>

// ---------------------------------------------------------------------------
// MI455X (gfx1250) implementation.
// f16 WMMA (v_wmma_f32_16x16x32_f16) with f32 accumulate for all GEMMs;
// f32 residual stream + LayerNorm; sequential scan in f32.
// GEMM is software-pipelined (ping-pong fragment registers, 2-stage unroll)
// so WMMAs overlap the next tile's global loads instead of full-waiting.
// ---------------------------------------------------------------------------

typedef __attribute__((ext_vector_type(16))) _Float16 v16h;
typedef __attribute__((ext_vector_type(8)))  _Float16 v8h;
typedef __attribute__((ext_vector_type(8)))  float    v8f;

#define DMODEL  512
#define DFF     2048
#define SEQLEN  2048
#define BATCHSZ 8
#define NLAYERS 4
#define NVOCAB  256
#define MROWS   (BATCHSZ * SEQLEN)   // 16384
#define LN_EPS  1e-5f

// ---------------------------------------------------------------------------
// Weight prep: transpose + convert f32 [K][N] -> f16 [N][K] (LDS-tiled).
// ---------------------------------------------------------------------------
__global__ __launch_bounds__(256)
void transpose_f32_to_f16(const float* __restrict__ W, _Float16* __restrict__ WT,
                          int K, int N) {
    __shared__ float tile[32][33];
    const int nt = blockIdx.x * 32;
    const int kt = blockIdx.y * 32;
    const int tx = threadIdx.x;   // 0..31
    const int ty = threadIdx.y;   // 0..7
    #pragma unroll
    for (int i = 0; i < 32; i += 8)
        tile[ty + i][tx] = W[(size_t)(kt + ty + i) * N + nt + tx];
    __syncthreads();
    #pragma unroll
    for (int i = 0; i < 32; i += 8)
        WT[(size_t)(nt + ty + i) * K + kt + tx] = (_Float16)tile[tx][ty + i];
}

// ---------------------------------------------------------------------------
// Embedding gather: x32[r,d] = emb[token[r], d]; also emit f16 copy for GEMM.
// ---------------------------------------------------------------------------
__global__ __launch_bounds__(256)
void embed_kernel(const long long* __restrict__ tokens, const float* __restrict__ emb,
                  float* __restrict__ x32, _Float16* __restrict__ xh) {
    const int idx = blockIdx.x * 256 + threadIdx.x;
    const int row = idx >> 9;          // / 512
    const int d   = idx & (DMODEL - 1);
    const long long t = tokens[row];
    const float v = emb[(size_t)t * DMODEL + d];
    x32[idx] = v;
    xh[idx]  = (_Float16)v;
}

// ---------------------------------------------------------------------------
// Sequential masked-decay scan. One thread per (batch, channel) pair.
// ---------------------------------------------------------------------------
__global__ __launch_bounds__(256)
void scan_kernel(const float* __restrict__ u, const long long* __restrict__ tokens,
                 const float* __restrict__ decay_p, _Float16* __restrict__ mh) {
    const int t = blockIdx.x * 256 + threadIdx.x;   // 0..4095
    const int b = t >> 9;
    const int d = t & (DMODEL - 1);
    const float decay = 1.0f / (1.0f + expf(-decay_p[0]));
    const long long* tk = tokens + (size_t)b * SEQLEN;
    const float*     up = u  + (size_t)b * SEQLEN * DMODEL + d;
    _Float16*        mp = mh + (size_t)b * SEQLEN * DMODEL + d;
    float mem = 0.0f;
    for (int s = 0; s < SEQLEN; ++s) {
        const float ui = up[(size_t)s * DMODEL];
        if (tk[s] != 0) mem = decay * mem + ui;    // msk*new + (1-msk)*mem
        mp[(size_t)s * DMODEL] = (_Float16)mem;
    }
}

// ---------------------------------------------------------------------------
// LayerNorm over d=512. One 256-thread block per row, LDS tree reduction.
// ---------------------------------------------------------------------------
__global__ __launch_bounds__(256)
void ln_kernel(const float* __restrict__ x, const float* __restrict__ g,
               const float* __restrict__ b, float* __restrict__ o32,
               _Float16* __restrict__ o16) {
    __shared__ float red[256];
    const int row = blockIdx.x;
    const int i = threadIdx.x;
    const float* xr = x + (size_t)row * DMODEL;
    const float a0 = xr[i];
    const float a1 = xr[i + 256];
    red[i] = a0 + a1;
    __syncthreads();
    for (int s = 128; s > 0; s >>= 1) { if (i < s) red[i] += red[i + s]; __syncthreads(); }
    const float mu = red[0] * (1.0f / (float)DMODEL);
    __syncthreads();
    const float d0 = a0 - mu, d1 = a1 - mu;
    red[i] = d0 * d0 + d1 * d1;
    __syncthreads();
    for (int s = 128; s > 0; s >>= 1) { if (i < s) red[i] += red[i + s]; __syncthreads(); }
    const float rstd = rsqrtf(red[0] * (1.0f / (float)DMODEL) + LN_EPS);
    const float y0 = d0 * rstd * g[i]       + b[i];
    const float y1 = d1 * rstd * g[i + 256] + b[i + 256];
    if (o32) { float* p = o32 + (size_t)row * DMODEL; p[i] = y0; p[i + 256] = y1; }
    if (o16) { _Float16* p = o16 + (size_t)row * DMODEL; p[i] = (_Float16)y0; p[i + 256] = (_Float16)y1; }
}

__global__ __launch_bounds__(256)
void f32_to_f16_kernel(const float* __restrict__ x, _Float16* __restrict__ y) {
    const int i = blockIdx.x * 256 + threadIdx.x;
    y[i] = (_Float16)x[i];
}

// ---------------------------------------------------------------------------
// Fragment loaders.
// A frag (16x32 f16): lane m holds 16 contiguous halves (k = 16*half + s).
// B frag (32x16 f16 from BT[N][K]): lane n holds 16 contiguous halves.
// ---------------------------------------------------------------------------
__device__ __forceinline__ v16h load_a_frag(const _Float16* p) {
    return *(const v16h*)p;
}

#define WMMA_F16(A_, B_, C_) \
    __builtin_amdgcn_wmma_f32_16x16x32_f16(false, (A_), false, (B_), (short)0, (C_), false, false)

// ---------------------------------------------------------------------------
// WMMA GEMM: C(MxN) = A(MxK, f16 row-major) * B(KxN given as BT[N][K] f16).
// Block: 256 threads = 8 waves arranged 4(M) x 2(N); block tile 128 x 128;
// wave tile 32(M) x 64(N) = 8 accumulators. K-loop is 2-stage software
// pipelined in steps of 64 (K % 64 == 0 for all calls). The final stage
// overfetches one 32-K tile (<=128B past the logical end); all GEMM input
// buffers are padded by 256B in the workspace to make that safe.
// EPI: 0 = bias -> f32; 1 = bias + in-place residual add -> f32;
//      2 = bias + exact GELU -> f16.
// ---------------------------------------------------------------------------
template <int EPI>
__global__ __launch_bounds__(256)
void gemm_wmma(const _Float16* __restrict__ A, const _Float16* __restrict__ BT,
               const float* __restrict__ bias, float* __restrict__ C32,
               _Float16* __restrict__ C16, int M, int N, int K) {
    const int lane  = threadIdx.x & 31;
    const int wave  = threadIdx.x >> 5;
    const int waveM = wave & 3;
    const int waveN = wave >> 2;
    const int row0  = blockIdx.x * 128 + waveM * 32;
    const int col0  = blockIdx.y * 128 + waveN * 64;

    // A fragment addressing (16-bit A 16x32 layout)
    const int am  = lane & 15;
    const int akh = (lane >> 4) * 16;
    const _Float16* ar0 = A + (size_t)(row0 + am) * K + akh;
    const _Float16* ar1 = ar0 + (size_t)16 * K;

    // B fragment addressing (BT is [N][K])
    const int bn  = lane & 15;
    const int bkb = (lane >> 4) * 16;
    const _Float16* br = BT + (size_t)(col0 + bn) * K + bkb;
    const size_t bstep = (size_t)16 * K;

    v8f acc[8];
    #pragma unroll
    for (int i = 0; i < 8; ++i) acc[i] = {};

    // Preload stage 0 (k = 0)
    v16h a0 = load_a_frag(ar0);
    v16h a1 = load_a_frag(ar1);
    v16h b0 = *(const v16h*)(br);
    v16h b1 = *(const v16h*)(br + bstep);
    v16h b2 = *(const v16h*)(br + 2 * bstep);
    v16h b3 = *(const v16h*)(br + 3 * bstep);

    for (int k0 = 0; k0 < K; k0 += 64) {
        // ---- issue loads for tile k0+32 (x-set) ----
        const v16h xa0 = load_a_frag(ar0 + k0 + 32);
        const v16h xa1 = load_a_frag(ar1 + k0 + 32);
        const _Float16* bx = br + k0 + 32;
        const v16h xb0 = *(const v16h*)(bx);
        const v16h xb1 = *(const v16h*)(bx + bstep);
        const v16h xb2 = *(const v16h*)(bx + 2 * bstep);
        const v16h xb3 = *(const v16h*)(bx + 3 * bstep);

        // ---- compute tile k0 ----
        acc[0] = WMMA_F16(a0, b0, acc[0]);
        acc[1] = WMMA_F16(a0, b1, acc[1]);
        acc[2] = WMMA_F16(a0, b2, acc[2]);
        acc[3] = WMMA_F16(a0, b3, acc[3]);
        acc[4] = WMMA_F16(a1, b0, acc[4]);
        acc[5] = WMMA_F16(a1, b1, acc[5]);
        acc[6] = WMMA_F16(a1, b2, acc[6]);
        acc[7] = WMMA_F16(a1, b3, acc[7]);

        // ---- issue loads for tile k0+64 (overfetch on last iter; padded) ----
        a0 = load_a_frag(ar0 + k0 + 64);
        a1 = load_a_frag(ar1 + k0 + 64);
        const _Float16* bc = br + k0 + 64;
        b0 = *(const v16h*)(bc);
        b1 = *(const v16h*)(bc + bstep);
        b2 = *(const v16h*)(bc + 2 * bstep);
        b3 = *(const v16h*)(bc + 3 * bstep);

        // ---- compute tile k0+32 ----
        acc[0] = WMMA_F16(xa0, xb0, acc[0]);
        acc[1] = WMMA_F16(xa0, xb1, acc[1]);
        acc[2] = WMMA_F16(xa0, xb2, acc[2]);
        acc[3] = WMMA_F16(xa0, xb3, acc[3]);
        acc[4] = WMMA_F16(xa1, xb0, acc[4]);
        acc[5] = WMMA_F16(xa1, xb1, acc[5]);
        acc[6] = WMMA_F16(xa1, xb2, acc[6]);
        acc[7] = WMMA_F16(xa1, xb3, acc[7]);
    }

    // C/D layout: VGPR i -> row (lane>=16 ? 8 : 0)+i, col = lane&15
    const int ncol  = lane & 15;
    const int rbase = (lane >> 4) * 8;
    #pragma unroll
    for (int mt = 0; mt < 2; ++mt) {
        #pragma unroll
        for (int nt = 0; nt < 4; ++nt) {
            #pragma unroll
            for (int i = 0; i < 8; ++i) {
                const int r = row0 + mt * 16 + rbase + i;
                const int c = col0 + nt * 16 + ncol;
                float v = acc[mt * 4 + nt][i] + bias[c];
                if (EPI == 0) {
                    C32[(size_t)r * N + c] = v;
                } else if (EPI == 1) {
                    const size_t o = (size_t)r * N + c;
                    C32[o] = C32[o] + v;
                } else {
                    const float ge = 0.5f * v * (1.0f + erff(v * 0.70710678118654752f));
                    C16[(size_t)r * N + c] = (_Float16)ge;
                }
            }
        }
    }
}

// ---------------------------------------------------------------------------
extern "C" void kernel_launch(void* const* d_in, const int* in_sizes, int n_in,
                              void* d_out, int out_size, void* d_ws, size_t ws_size,
                              hipStream_t stream) {
    (void)in_sizes; (void)n_in; (void)out_size; (void)ws_size;

    const long long* tokens = (const long long*)d_in[0];
    const float* emb     = (const float*)d_in[1];
    const float* W_upd   = (const float*)d_in[2];
    const float* b_upd   = (const float*)d_in[3];
    const float* W_f     = (const float*)d_in[4];
    const float* b_f     = (const float*)d_in[5];
    const float* decay_p = (const float*)d_in[6];
    const float* norm_g  = (const float*)d_in[7];
    const float* norm_b  = (const float*)d_in[8];
    const float* ln_g    = (const float*)d_in[9];
    const float* ln_b    = (const float*)d_in[10];
    const float* W1      = (const float*)d_in[11];
    const float* b1      = (const float*)d_in[12];
    const float* W2      = (const float*)d_in[13];
    const float* b2      = (const float*)d_in[14];
    const float* head_W  = (const float*)d_in[15];
    const float* head_b  = (const float*)d_in[16];
    float* out = (float*)d_out;

    // Workspace carve-up (~186 MB), 256B aligned; +256B pad on every buffer
    // that feeds the GEMM (A or BT) to absorb the pipeline overfetch.
    size_t off = 0;
    char* base = (char*)d_ws;
    auto take = [&](size_t bytes) -> char* {
        char* p = base + off;
        off += (bytes + 511) & ~(size_t)255;   // >= bytes + 256 of slack
        return p;
    };
    _Float16* wupdT = (_Float16*)take((size_t)DMODEL * DMODEL * 2);
    _Float16* wfT   = (_Float16*)take((size_t)DMODEL * DMODEL * 2);
    _Float16* w1T   = (_Float16*)take((size_t)NLAYERS * DMODEL * DFF * 2);
    _Float16* w2T   = (_Float16*)take((size_t)NLAYERS * DFF * DMODEL * 2);
    _Float16* whT   = (_Float16*)take((size_t)DMODEL * NVOCAB * 2);
    float*    x32   = (float*)   take((size_t)MROWS * DMODEL * 4);
    float*    u32   = (float*)   take((size_t)MROWS * DMODEL * 4);
    _Float16* xh    = (_Float16*)take((size_t)MROWS * DMODEL * 2);
    _Float16* mh    = (_Float16*)take((size_t)MROWS * DMODEL * 2);  // mems; reused as LN-out
    _Float16* h2    = (_Float16*)take((size_t)MROWS * DFF * 2);

    const dim3 tb(32, 8);

    // --- weight prep (f32 [K][N] -> f16 [N][K]) ---
    transpose_f32_to_f16<<<dim3(DMODEL / 32, DMODEL / 32), tb, 0, stream>>>(W_upd, wupdT, DMODEL, DMODEL);
    transpose_f32_to_f16<<<dim3(DMODEL / 32, DMODEL / 32), tb, 0, stream>>>(W_f,   wfT,   DMODEL, DMODEL);
    for (int i = 0; i < NLAYERS; ++i) {
        transpose_f32_to_f16<<<dim3(DFF / 32, DMODEL / 32), tb, 0, stream>>>(
            W1 + (size_t)i * DMODEL * DFF, w1T + (size_t)i * DMODEL * DFF, DMODEL, DFF);
        transpose_f32_to_f16<<<dim3(DMODEL / 32, DFF / 32), tb, 0, stream>>>(
            W2 + (size_t)i * DFF * DMODEL, w2T + (size_t)i * DFF * DMODEL, DFF, DMODEL);
    }
    transpose_f32_to_f16<<<dim3(NVOCAB / 32, DMODEL / 32), tb, 0, stream>>>(head_W, whT, DMODEL, NVOCAB);

    // --- embedding gather ---
    embed_kernel<<<(MROWS * DMODEL) / 256, 256, 0, stream>>>(tokens, emb, x32, xh);

    // --- fast memory: u = x @ W_upd + b_upd ---
    gemm_wmma<0><<<dim3(MROWS / 128, DMODEL / 128), 256, 0, stream>>>(
        xh, wupdT, b_upd, u32, nullptr, MROWS, DMODEL, DMODEL);

    // --- masked decay scan -> mems (f16) ---
    scan_kernel<<<(BATCHSZ * DMODEL) / 256, 256, 0, stream>>>(u32, tokens, decay_p, mh);

    // --- x += mems @ W_f + b_f ---
    gemm_wmma<1><<<dim3(MROWS / 128, DMODEL / 128), 256, 0, stream>>>(
        mh, wfT, b_f, x32, nullptr, MROWS, DMODEL, DMODEL);

    // --- x = LN(x) (in place, f32) ---
    ln_kernel<<<MROWS, 256, 0, stream>>>(x32, norm_g, norm_b, x32, nullptr);

    // --- 4 MLP layers ---
    for (int i = 0; i < NLAYERS; ++i) {
        ln_kernel<<<MROWS, 256, 0, stream>>>(x32, ln_g + i * DMODEL, ln_b + i * DMODEL, nullptr, mh);
        gemm_wmma<2><<<dim3(MROWS / 128, DFF / 128), 256, 0, stream>>>(
            mh, w1T + (size_t)i * DMODEL * DFF, b1 + (size_t)i * DFF,
            nullptr, h2, MROWS, DFF, DMODEL);
        gemm_wmma<1><<<dim3(MROWS / 128, DMODEL / 128), 256, 0, stream>>>(
            h2, w2T + (size_t)i * DFF * DMODEL, b2 + (size_t)i * DMODEL,
            x32, nullptr, MROWS, DMODEL, DFF);
    }

    // --- head: out = x @ head_W + head_b ---
    f32_to_f16_kernel<<<(MROWS * DMODEL) / 256, 256, 0, stream>>>(x32, xh);
    gemm_wmma<0><<<dim3(MROWS / 128, NVOCAB / 128), 256, 0, stream>>>(
        xh, whT, head_b, out, nullptr, MROWS, NVOCAB, DMODEL);
}
